// AFTFull_24979529793898
// MI455X (gfx1250) — matrix operand order repaired and checked by reading, not compiled, not run on hardware
//
#include <hip/hip_runtime.h>
#include <hip/hip_bf16.h>
#include <math.h>

// ---------------------------------------------------------------------------
// AFT-Full for MI455X (gfx1250, wave32): all GEMMs on v_wmma_f32_16x16x32_bf16.
// Projection GEMM: B staged to LDS manually (global->VGPR->ds_store).
// AFT kernel: B block moved by the Tensor Data Mover (tensor_load_to_lds) with
// descriptor-driven LDS padding for conflict-free ds_load_b128.
// ---------------------------------------------------------------------------

typedef __bf16 bf16_t;
typedef __attribute__((ext_vector_type(16))) __bf16 v16bf;
typedef __attribute__((ext_vector_type(8)))  __bf16 v8bf;
typedef __attribute__((ext_vector_type(8)))  float  v8f;
typedef __attribute__((ext_vector_type(4)))  unsigned int v4u;
typedef __attribute__((ext_vector_type(8)))  int  v8i;
typedef __attribute__((ext_vector_type(4)))  int  v4i;

#define SEQ   2048
#define DM    1024
#define NH    16
#define DH    64
#define BATCH 4
#define BT    (BATCH * SEQ)   // 8192 tokens

// LDS row stride for staged B tiles: 32 payload bf16 + 8 pad (64B + 16B).
// 80B stride -> lane n's b128 read starts at bank (20*n mod 64): 16 distinct
// 4-bank groups across the 16 lanes => conflict-free ds_load_b128.
// TDM equivalent: pad_interval = 16 DWORDs (code 3), pad_amount = 4 DW (code 3).
#define LDSB_STRIDE 40

// ---- WMMA operand loaders (wave32, 16x16x32 bf16 layouts per CDNA5 ISA) ----
__device__ __forceinline__ v16bf load_a_tile(const bf16_t* A, int lda) {
  const int lane = threadIdx.x & 31;
  const int half = lane >> 4;
  const int m    = lane & 15;
  const bf16_t* p = A + (size_t)m * lda + half * 8;
  v8bf lo = *(const v8bf*)(p);        // K = 8*half .. +7
  v8bf hi = *(const v8bf*)(p + 16);   // K = 16+8*half .. +7
  v16bf r;
#pragma unroll
  for (int i = 0; i < 8; ++i) { r[i] = lo[i]; r[8 + i] = hi[i]; }
  return r;
}

__device__ __forceinline__ v16bf load_b_lds(const bf16_t* Brow0) {
  const int lane = threadIdx.x & 31;
  const int half = lane >> 4;
  const int n    = lane & 15;
  const bf16_t* p = Brow0 + n * LDSB_STRIDE + half * 8;
  v8bf lo = *(const v8bf*)(p);        // K = 8*half .. +7
  v8bf hi = *(const v8bf*)(p + 16);   // K = 16+8*half .. +7
  v16bf r;
#pragma unroll
  for (int i = 0; i < 8; ++i) { r[i] = lo[i]; r[8 + i] = hi[i]; }
  return r;
}

__device__ __forceinline__ v8f zero8() {
  v8f z = {0.f, 0.f, 0.f, 0.f, 0.f, 0.f, 0.f, 0.f};
  return z;
}

// ---------------------------------------------------------------------------
// Projection GEMM: C[M,N] = A[M,K](bf16) @ Wt[N,K]^T(bf16) + bias
// block = 128 threads (4 waves); block tile 128(M) x 64(N); wave = 32x64.
// ---------------------------------------------------------------------------
__global__ void __launch_bounds__(128)
gemm_bf16_proj(const bf16_t* __restrict__ A,
               const bf16_t* __restrict__ Wt,
               const float*  __restrict__ bias,
               float*        __restrict__ outF,
               bf16_t*       __restrict__ ekv2,
               int M, int N, int K, int mode)
{
  __shared__ __attribute__((aligned(128))) bf16_t sB[64 * LDSB_STRIDE];

  const int tid  = threadIdx.x;
  const int wave = tid >> 5;
  const int m0   = blockIdx.x * 128 + wave * 32;
  const int n0   = blockIdx.y * 64;
  const int lane = tid & 31;
  const int half = lane >> 4;
  const int ln   = lane & 15;

  v8f acc[2][4];
#pragma unroll
  for (int mt = 0; mt < 2; ++mt)
#pragma unroll
    for (int j = 0; j < 4; ++j) acc[mt][j] = zero8();

  for (int k0 = 0; k0 < K; k0 += 32) {
    // cooperative stage of B block: rows n0..n0+63, cols k0..k0+31
    {
      const int r    = tid >> 1;
      const int part = tid & 1;
      const bf16_t* src = Wt + (size_t)(n0 + r) * K + k0 + part * 16;
      v8bf c0 = *(const v8bf*)(src);
      v8bf c1 = *(const v8bf*)(src + 8);
      bf16_t* dst = sB + r * LDSB_STRIDE + part * 16;
      *(v8bf*)(dst)     = c0;
      *(v8bf*)(dst + 8) = c1;
    }
    __syncthreads();

    v16bf a0 = load_a_tile(A + (size_t)m0 * K + k0, K);
    v16bf a1 = load_a_tile(A + (size_t)(m0 + 16) * K + k0, K);
    if (k0 + 32 < K) {
      __builtin_prefetch(A + (size_t)m0 * K + k0 + 32, 0, 1);
      __builtin_prefetch(A + (size_t)(m0 + 16) * K + k0 + 32, 0, 1);
    }

#pragma unroll
    for (int j = 0; j < 4; ++j) {
      v16bf b = load_b_lds(sB + (16 * j) * LDSB_STRIDE);
      acc[0][j] = __builtin_amdgcn_wmma_f32_16x16x32_bf16(
          false, a0, false, b, (short)0, acc[0][j], false, false);
      acc[1][j] = __builtin_amdgcn_wmma_f32_16x16x32_bf16(
          false, a1, false, b, (short)0, acc[1][j], false, false);
    }
    __syncthreads();
  }

#pragma unroll
  for (int mt = 0; mt < 2; ++mt) {
#pragma unroll
    for (int j = 0; j < 4; ++j) {
#pragma unroll
      for (int i = 0; i < 8; ++i) {
        const int m = m0 + mt * 16 + i + 8 * half;
        const int n = n0 + 16 * j + ln;
        const float v = acc[mt][j][i] + bias[n];
        if (mode == 0) {
          outF[(size_t)m * N + n] = 1.f / (1.f + __expf(-v));
        } else if (mode == 3) {
          outF[(size_t)m * N + n] = v;
        } else {
          const int b = m >> 11, s = m & (SEQ - 1);
          const int h = n >> 6,  d = n & (DH - 1);
          const size_t base = (size_t)(b * NH + h) * 128 * SEQ;
          if (mode == 1) {
            ekv2[base + (size_t)(64 + d) * SEQ + s] = (bf16_t)__expf(v);
          } else { // mode 2: eKV = exp(K) * V
            const float ek = (float)ekv2[base + (size_t)(64 + d) * SEQ + s];
            ekv2[base + (size_t)d * SEQ + s] = (bf16_t)(ek * v);
          }
        }
      }
    }
  }
}

// ---------------------------------------------------------------------------
// AFT mixing: per (b,h)  numer = ew @ eKV,  denom = ew @ eK  (K = SEQ = 2048)
// B block (128 x 32 bf16 tile of the [128, SEQ] ekv2 slice) is DMA'd to LDS by
// the Tensor Data Mover. The D# requests padding (16 DW interval, 4 DW amount)
// so the LDS image has the conflict-free LDSB_STRIDE row pitch.
// ---------------------------------------------------------------------------
__global__ void __launch_bounds__(128)
aft_attn_kernel(const bf16_t* __restrict__ ewb,   // [SEQ, SEQ] bf16
                const bf16_t* __restrict__ ekv2,  // [B*H, 128, SEQ] bf16
                const float*  __restrict__ qsig,  // [BT, DM] f32 (sigmoid(Q))
                bf16_t*       __restrict__ attn)  // [BT, DM] bf16
{
  __shared__ __attribute__((aligned(128))) bf16_t sB[128 * LDSB_STRIDE];

  const int tid  = threadIdx.x;
  const int wave = tid >> 5;
  const int t0   = blockIdx.x * 64 + wave * 16;
  const int bh   = blockIdx.y;
  const int b    = bh >> 4, h = bh & 15;
  const int lane = tid & 31;
  const int half = lane >> 4;
  const int ln   = lane & 15;

  const bf16_t* Bt = ekv2 + (size_t)bh * 128 * SEQ;

  // ---- Tensor DMA descriptor (constant parts), per CDNA5 ISA ch.8 ----
  const unsigned int lds_off = (unsigned int)(size_t)(&sB[0]); // LDS byte addr
  // group1: data_size=2B(code1), pad_enable, pad_interval=16DW(3), pad_amount=4DW(3)
  const unsigned int g1w0 = (1u << 16) | (1u << 20) | (3u << 22) | (3u << 25);
  v8i g1;
  g1[0] = (int)g1w0;
  g1[1] = (int)((unsigned)(SEQ & 0xFFFF) << 16);          // tensor_dim0 lo16 @ [63:48]
  g1[2] = (int)(((unsigned)SEQ >> 16) | (128u << 16));    // dim0 hi16, tensor_dim1 lo16
  g1[3] = (int)(32u << 16);                               // tensor_dim1 hi16=0, tile_dim0=32
  g1[4] = (int)128u;                                      // tile_dim1=128, tile_dim2=0
  g1[5] = (int)(unsigned)SEQ;                             // tensor_dim0_stride lo32
  g1[6] = 0;                                              // stride hi16, dim1_stride lo16
  g1[7] = 0;
  v4i gz4; gz4[0] = 0; gz4[1] = 0; gz4[2] = 0; gz4[3] = 0; // groups 2/3 unused (2D)
  v8i gz8;
#pragma unroll
  for (int i = 0; i < 8; ++i) gz8[i] = 0;                  // extra group (6-arg form)

  v8f acc[8];
#pragma unroll
  for (int j = 0; j < 8; ++j) acc[j] = zero8();

  for (int k0 = 0; k0 < SEQ; k0 += 32) {
    // ---- TDM: move 128x32 bf16 tile (8KB) into LDS; wave 0 issues ----
    if (wave == 0) {
      const unsigned long long ga =
          (unsigned long long)(size_t)Bt + (unsigned long long)k0 * 2ull;
      v4u g0;
      g0[0] = 1u;                                   // count=1, user descriptor
      g0[1] = lds_off;                              // lds_addr @ [63:32]
      g0[2] = (unsigned int)ga;                     // global_addr lo32
      g0[3] = (unsigned int)((ga >> 32) & 0x1FFFFFFull) | (2u << 30); // type=2
      __builtin_amdgcn_tensor_load_to_lds(g0, g1, gz4, gz4, gz8, 0);
      __builtin_amdgcn_s_wait_tensorcnt(0);
    }
    __syncthreads();

    v16bf a = load_a_tile(ewb + (size_t)t0 * SEQ + k0, SEQ);
    if (k0 + 32 < SEQ)
      __builtin_prefetch(ewb + (size_t)t0 * SEQ + k0 + 32, 0, 1);

#pragma unroll
    for (int j = 0; j < 8; ++j) {
      v16bf bm = load_b_lds(sB + (16 * j) * LDSB_STRIDE);
      acc[j] = __builtin_amdgcn_wmma_f32_16x16x32_bf16(
          false, a, false, bm, (short)0, acc[j], false, false);
    }
    __syncthreads();
  }

#pragma unroll
  for (int j = 0; j < 4; ++j) {
#pragma unroll
    for (int i = 0; i < 8; ++i) {
      const int t = t0 + i + 8 * half;
      const int d = 16 * j + ln;
      const float numer = acc[j][i];
      const float denom = acc[4 + j][i];
      const size_t idx = (size_t)(b * SEQ + t) * DM + h * DH + d;
      attn[idx] = (bf16_t)(qsig[idx] * numer / denom);
    }
  }
}

// ---------------------------------------------------------------------------
// Elementwise prep
// ---------------------------------------------------------------------------
__global__ void cvt_f32_bf16(const float* __restrict__ src,
                             bf16_t* __restrict__ dst, int n) {
  for (int i = blockIdx.x * blockDim.x + threadIdx.x; i < n;
       i += gridDim.x * blockDim.x)
    dst[i] = (bf16_t)src[i];
}

__global__ void exp_f32_bf16(const float* __restrict__ src,
                             bf16_t* __restrict__ dst, int n) {
  for (int i = blockIdx.x * blockDim.x + threadIdx.x; i < n;
       i += gridDim.x * blockDim.x)
    dst[i] = (bf16_t)__expf(src[i]);
}

// ---------------------------------------------------------------------------
extern "C" void kernel_launch(void* const* d_in, const int* in_sizes, int n_in,
                              void* d_out, int out_size, void* d_ws, size_t ws_size,
                              hipStream_t stream) {
  const float* x     = (const float*)d_in[0];
  const float* Wq    = (const float*)d_in[2];
  const float* bq    = (const float*)d_in[3];
  const float* Wk    = (const float*)d_in[4];
  const float* bk    = (const float*)d_in[5];
  const float* Wv    = (const float*)d_in[6];
  const float* bv    = (const float*)d_in[7];
  const float* Wo    = (const float*)d_in[8];
  const float* bo    = (const float*)d_in[9];
  const float* wbias = (const float*)d_in[10];

  // workspace carve-up (~112 MB total)
  char* w = (char*)d_ws;
  size_t off = 0;
  bf16_t* xb   = (bf16_t*)(w + off); off += (size_t)BT * DM * 2;      // 16 MB
  bf16_t* wqb  = (bf16_t*)(w + off); off += (size_t)DM * DM * 2;      //  2 MB
  bf16_t* wkb  = (bf16_t*)(w + off); off += (size_t)DM * DM * 2;
  bf16_t* wvb  = (bf16_t*)(w + off); off += (size_t)DM * DM * 2;
  bf16_t* wob  = (bf16_t*)(w + off); off += (size_t)DM * DM * 2;
  bf16_t* ewb  = (bf16_t*)(w + off); off += (size_t)SEQ * SEQ * 2;    //  8 MB
  float*  qsig = (float*)(w + off);  off += (size_t)BT * DM * 4;      // 32 MB
  bf16_t* ekv2 = (bf16_t*)(w + off); off += (size_t)BATCH * NH * 128 * SEQ * 2; // 32 MB
  bf16_t* attn = (bf16_t*)(w + off); off += (size_t)BT * DM * 2;      // 16 MB
  (void)ws_size; (void)n_in; (void)in_sizes; (void)out_size;

  // 1) precision conversion + exp(wbias)
  cvt_f32_bf16<<<4096, 256, 0, stream>>>(x, xb, BT * DM);
  cvt_f32_bf16<<<2048, 256, 0, stream>>>(Wq, wqb, DM * DM);
  cvt_f32_bf16<<<2048, 256, 0, stream>>>(Wk, wkb, DM * DM);
  cvt_f32_bf16<<<2048, 256, 0, stream>>>(Wv, wvb, DM * DM);
  cvt_f32_bf16<<<2048, 256, 0, stream>>>(Wo, wob, DM * DM);
  exp_f32_bf16<<<4096, 256, 0, stream>>>(wbias, ewb, SEQ * SEQ);

  // 2) projections (WMMA bf16 GEMMs with fused epilogues)
  dim3 g1(BT / 128, DM / 64);   // (64, 16)
  gemm_bf16_proj<<<g1, 128, 0, stream>>>(xb, wqb, bq, qsig, nullptr, BT, DM, DM, 0);
  gemm_bf16_proj<<<g1, 128, 0, stream>>>(xb, wkb, bk, nullptr, ekv2, BT, DM, DM, 1);
  gemm_bf16_proj<<<g1, 128, 0, stream>>>(xb, wvb, bv, nullptr, ekv2, BT, DM, DM, 2);

  // 3) AFT mixing: TDM-fed numer/denom GEMMs + sigmoid(Q)*numer/denom
  dim3 g2(SEQ / 64, BATCH * NH);   // (32, 64)
  aft_attn_kernel<<<g2, 128, 0, stream>>>(ewb, ekv2, qsig, attn);

  // 4) output projection -> f32 d_out
  gemm_bf16_proj<<<g1, 128, 0, stream>>>(attn, wob, bo, (float*)d_out, nullptr, BT, DM, DM, 3);
}
